// Attention_83597243449567
// MI455X (gfx1250) — matrix operand order repaired and checked by reading, not compiled, not run on hardware
//
#include <hip/hip_runtime.h>
#include <hip/hip_bf16.h>

// ---------------------------------------------------------------------------
// MI455X (gfx1250, wave32) attention forward:
//   qkv = x @ w_qkv^T          (bf16x3-split WMMA GEMM; epilogue stores q
//                               token-major and k,v transposed per head so the
//                               attention WMMA B-operands are contiguous)
//   S   = q @ k  (non-standard, contracts q-features vs k-tokens)
//   P   = softmax over columns of S  (== rows after the reference transpose)
//   ctx = P @ v                 (WMMA, P fed straight from S accumulators)
//   out = ctx @ w_proj^T + b_proj   (bf16x3-split WMMA GEMM)
// All matmuls use V_WMMA_F32_16X16X32_BF16 with 3-term bf16 splitting
// (a_hi*b_hi + a_hi*b_lo + a_lo*b_hi, f32 accumulate) for near-fp32 accuracy
// at bf16 matrix-core rates.
// ---------------------------------------------------------------------------

#define DIMC   2048
#define NHEAD  8
#define HDIM   256
#define BATCH  64
#define SEQ    256
#define MTOT   (BATCH * SEQ)   // 16384 tokens
#define QKV_N  (3 * DIMC)      // 6144

// qkv plane internal layout (element offsets within one plane of 100663296):
//   [0,              33554432)  q   : [m = b*256+tok][feat 0..2047]
//   [33554432,       67108864)  kT  : [b*8+h][feature j][token t]
//   [67108864,      100663296)  vT  : [b*8+h][feature d][token t]
#define SECT  ((size_t)MTOT * DIMC)        // 33554432
#define HEADQ ((size_t)HDIM * SEQ)         // 65536 per head in kT/vT

typedef __attribute__((ext_vector_type(16))) __bf16         v16bf;
typedef __attribute__((ext_vector_type(8)))  float          v8f;
typedef __attribute__((ext_vector_type(16))) unsigned short v16u;
typedef __attribute__((ext_vector_type(8)))  unsigned short u16x8;

static __device__ __forceinline__ v16bf as_bf16(v16u u) {
    union { v16u u; v16bf b; } x; x.u = u; return x.b;
}
static __device__ __forceinline__ unsigned short f2bf(float f) {
    unsigned u = __float_as_uint(f);
    u += 0x7FFFu + ((u >> 16) & 1u);          // round-to-nearest-even
    return (unsigned short)(u >> 16);
}
static __device__ __forceinline__ float bf2f(unsigned short h) {
    return __uint_as_float(((unsigned)h) << 16);
}
static __device__ __forceinline__ void splitf(float f, unsigned short& hi, unsigned short& lo) {
    hi = f2bf(f);
    lo = f2bf(f - bf2f(hi));
}

// D = A*B + C  (bf16 in, f32 accumulate)
static __device__ __forceinline__ v8f wmma_bf(v16u a, v16u b, v8f c) {
    return __builtin_amdgcn_wmma_f32_16x16x32_bf16(
        false, as_bf16(a), false, as_bf16(b), (short)0, c, false, false);
}

// Assemble a 16-lane-value operand from two contiguous 8x u16 chunks (b128 loads).
static __device__ __forceinline__ v16u ld_op16(const unsigned short* p0,
                                               const unsigned short* p1) {
    u16x8 a = *(const u16x8*)p0;
    u16x8 b = *(const u16x8*)p1;
    v16u r;
#pragma unroll
    for (int i = 0; i < 8; ++i) { r[i] = a[i]; r[8 + i] = b[i]; }
    return r;
}

// ---------------------------------------------------------------------------
// Split fp32 -> (bf16_hi, bf16_lo) planes (weights, once per launch).
// ---------------------------------------------------------------------------
__global__ void split_f32_kernel(const float* __restrict__ src,
                                 unsigned short* __restrict__ hi,
                                 unsigned short* __restrict__ lo, int n) {
    int i = blockIdx.x * 256 + threadIdx.x;
    if (i < n) {
        unsigned short h, l;
        splitf(src[i], h, l);
        hi[i] = h; lo[i] = l;
    }
}

// ---------------------------------------------------------------------------
// C[M,N] = A[M,K] @ W[N,K]^T  with bf16x3 split.
// A either fp32 (split on the fly) or pre-split hi/lo planes.
// OUT_MODE 0: qkv scatter -> q token-major, kT/vT per-head feature-major planes
// OUT_MODE 1: fp32 with bias added
// Block: 256 threads = 8 waves. Block tile 128(M) x 128(N); wave -> 16 rows,
// 8 C tiles (A operand reused 8x). W-plane rows read contiguously (b128).
// ---------------------------------------------------------------------------
template <bool A_IS_F32, int OUT_MODE>
__global__ void __launch_bounds__(256, 1)
gemm_bf16x3_kernel(const float* __restrict__ Af,
                   const unsigned short* __restrict__ Ahi,
                   const unsigned short* __restrict__ Alo,
                   const unsigned short* __restrict__ Whi,
                   const unsigned short* __restrict__ Wlo,
                   const float* __restrict__ bias,
                   float* __restrict__ Of,
                   unsigned short* __restrict__ Ohi,
                   unsigned short* __restrict__ Olo,
                   int M, int N, int K) {
    const int lane = threadIdx.x & 31;
    const int wave = threadIdx.x >> 5;
    const int l15  = lane & 15;
    const int up   = lane >> 4;        // 0 for lanes 0-15, 1 for 16-31
    const int akb  = up * 8;           // A-operand K sub-offset
    const int bkb  = up * 16;          // unused; B operand also uses akb
    const int arow = blockIdx.y * 128 + wave * 16 + l15;
    const int col0 = blockIdx.x * 128;

    v8f z = {};
    v8f acc[8] = {z, z, z, z, z, z, z, z};

    for (int kc = 0; kc < K; kc += 32) {
        v16u ah, al;
        if constexpr (A_IS_F32) {
            const float* ap = Af + (size_t)arow * K + kc + akb;
            if (kc + 32 < K) __builtin_prefetch(ap + 32, 0, 1);
#pragma unroll
            for (int e = 0; e < 8; ++e) {
                unsigned short h, l;
                splitf(ap[e], h, l);      ah[e] = h;     al[e] = l;
                splitf(ap[16 + e], h, l); ah[8 + e] = h; al[8 + e] = l;
            }
        } else {
            const unsigned short* ph = Ahi + (size_t)arow * K + kc + akb;
            const unsigned short* pl = Alo + (size_t)arow * K + kc + akb;
            if (kc + 32 < K) {
                __builtin_prefetch(ph + 32, 0, 1);
                __builtin_prefetch(pl + 32, 0, 1);
            }
            ah = ld_op16(ph, ph + 16);
            al = ld_op16(pl, pl + 16);
        }
#pragma unroll
        for (int nt = 0; nt < 8; ++nt) {
            const size_t boff = (size_t)(col0 + nt * 16 + l15) * K + kc + akb;
            v16u bh = ld_op16(Whi + boff, Whi + boff + 16);
            v16u bl = ld_op16(Wlo + boff, Wlo + boff + 16);
            acc[nt] = wmma_bf(ah, bh, acc[nt]);   // hi*hi
            acc[nt] = wmma_bf(ah, bl, acc[nt]);   // hi*lo
            acc[nt] = wmma_bf(al, bh, acc[nt]);   // lo*hi
        }
    }

#pragma unroll
    for (int nt = 0; nt < 8; ++nt) {
#pragma unroll
        for (int r = 0; r < 8; ++r) {
            const int orow = blockIdx.y * 128 + wave * 16 + r + up * 8;
            const int ocol = col0 + nt * 16 + l15;
            const float v  = acc[nt][r];
            if constexpr (OUT_MODE == 1) {
                Of[(size_t)orow * N + ocol] = v + bias[ocol];
            } else {
                // qkv scatter: q row-major; k,v transposed to [head][feat][tok]
                size_t idx;
                if (ocol < DIMC) {
                    idx = (size_t)orow * DIMC + ocol;             // q
                } else {
                    const int sec = ocol >> 11;                   // 1=k, 2=v
                    const int hf  = ocol & (DIMC - 1);
                    const int hh  = hf >> 8;
                    const int f   = hf & (HDIM - 1);
                    const int bb  = orow >> 8;
                    const int t   = orow & (SEQ - 1);
                    idx = (size_t)sec * SECT +
                          ((size_t)(bb * NHEAD + hh) * HDIM + f) * SEQ + t;
                }
                unsigned short h, l;
                splitf(v, h, l);
                Ohi[idx] = h;
                Olo[idx] = l;
            }
        }
    }
}

// ---------------------------------------------------------------------------
// Attention: 1024 blocks = (batch*head) x 2 j-halves; 128 threads = 4 waves,
// each wave 2 j-tiles. With kT/vT feature-major planes every WMMA operand is
// two contiguous b128 loads; P for the second GEMM comes straight from the
// S accumulators (C-layout == A-operand layout, lane-exact).
//   S[i,j] = sum_t q[i,t] * kT[j,t]
//   softmax over i for fixed j  == in-lane reduction + shfl_xor(16)
//   O[j,d] = sum_i P[j,i] * vT[d,i]   (d in 4 quarters: O[4] = 32 VGPRs)
// ---------------------------------------------------------------------------
__global__ void __launch_bounds__(128, 1)
attn_kernel(const unsigned short* __restrict__ qkv_hi,
            const unsigned short* __restrict__ qkv_lo,
            unsigned short* __restrict__ ctx_hi,
            unsigned short* __restrict__ ctx_lo) {
    const int bhid = blockIdx.x >> 1;   // b*8 + h
    const int jblk = blockIdx.x & 1;    // which 128-wide j half
    const int b  = bhid >> 3;
    const int h  = bhid & 7;
    const int lane = threadIdx.x & 31;
    const int wave = threadIdx.x >> 5;  // 0..3
    const int l15  = lane & 15;
    const int up   = lane >> 4;
    const int akb  = up * 8;
    const int bkb  = up * 16;

    const unsigned short* kT_hi = qkv_hi + SECT     + (size_t)bhid * HEADQ;
    const unsigned short* kT_lo = qkv_lo + SECT     + (size_t)bhid * HEADQ;
    const unsigned short* vT_hi = qkv_hi + 2 * SECT + (size_t)bhid * HEADQ;
    const unsigned short* vT_lo = qkv_lo + 2 * SECT + (size_t)bhid * HEADQ;
    const size_t qrow0 = (size_t)b * SEQ;           // token row base in q plane

    v8f z = {};

    for (int jrep = 0; jrep < 2; ++jrep) {
        const int j0 = jblk * 128 + (wave * 2 + jrep) * 16;

        // ---- Phase 1: S tiles (16 i-tiles x this j-tile), K over features t
        v8f S[16];
#pragma unroll
        for (int it = 0; it < 16; ++it) S[it] = z;

        for (int kc = 0; kc < HDIM; kc += 32) {
            // B operand: kT row (j0+l15), tokens t = kc + akb + {0..7, 16..23}
            const size_t kb = (size_t)(j0 + l15) * SEQ + kc + akb;
            v16u bh_ = ld_op16(kT_hi + kb, kT_hi + kb + 16);
            v16u bl_ = ld_op16(kT_lo + kb, kT_lo + kb + 16);
#pragma unroll
            for (int it = 0; it < 16; ++it) {
                const size_t aoff =
                    (qrow0 + it * 16 + l15) * DIMC + h * HDIM + kc + akb;
                v16u ah = ld_op16(qkv_hi + aoff, qkv_hi + aoff + 16);
                v16u al = ld_op16(qkv_lo + aoff, qkv_lo + aoff + 16);
                S[it] = wmma_bf(ah, bh_, S[it]);
                S[it] = wmma_bf(ah, bl_, S[it]);
                S[it] = wmma_bf(al, bh_, S[it]);
            }
        }

        // ---- Softmax over i (reference transposes then softmaxes last axis,
        //      i.e. over columns of S; each lane owns one column-half).
        float m = -3.0e38f;
#pragma unroll
        for (int it = 0; it < 16; ++it)
#pragma unroll
            for (int r = 0; r < 8; ++r) m = fmaxf(m, S[it][r]);
        m = fmaxf(m, __shfl_xor(m, 16, 32));
        float sum = 0.0f;
#pragma unroll
        for (int it = 0; it < 16; ++it)
#pragma unroll
            for (int r = 0; r < 8; ++r) {
                float e = __expf(S[it][r] - m);
                S[it][r] = e;
                sum += e;
            }
        sum += __shfl_xor(sum, 16, 32);
        const float inv = 1.0f / sum;

        // ---- Phase 2: O = P @ v, d in 4 quarters to bound register pressure.
        for (int dh = 0; dh < 4; ++dh) {
            v8f O[4];
#pragma unroll
            for (int d4 = 0; d4 < 4; ++d4) O[d4] = z;

            for (int tt = 0; tt < 8; ++tt) {   // K window: i in [32*tt, 32*tt+32)
                // P tiles in C-layout are already exactly the A-operand layout:
                v16u ah, al;
#pragma unroll
                for (int e = 0; e < 8; ++e) {
                    unsigned short hh, ll;
                    splitf(S[2 * tt][e] * inv, hh, ll);     ah[e] = hh;     al[e] = ll;
                    splitf(S[2 * tt + 1][e] * inv, hh, ll); ah[8 + e] = hh; al[8 + e] = ll;
                }
#pragma unroll
                for (int d4 = 0; d4 < 4; ++d4) {
                    const int dcol = (dh * 4 + d4) * 16 + l15;
                    // B operand: vT row (dcol), tokens i = tt*32 + akb + {0..7, 16..23}
                    const size_t vb = (size_t)dcol * SEQ + tt * 32 + akb;
                    v16u bh2 = ld_op16(vT_hi + vb, vT_hi + vb + 16);
                    v16u bl2 = ld_op16(vT_lo + vb, vT_lo + vb + 16);
                    O[d4] = wmma_bf(ah, bh2, O[d4]);
                    O[d4] = wmma_bf(ah, bl2, O[d4]);
                    O[d4] = wmma_bf(al, bh2, O[d4]);
                }
            }

            // ctx[b, token=j, h*256 + d] as split planes for the final GEMM.
#pragma unroll
            for (int d4 = 0; d4 < 4; ++d4) {
#pragma unroll
                for (int r = 0; r < 8; ++r) {
                    const int tok  = j0 + r + up * 8;
                    const int feat = h * HDIM + (dh * 4 + d4) * 16 + l15;
                    const size_t idx = ((size_t)b * SEQ + tok) * DIMC + feat;
                    unsigned short hh, ll;
                    splitf(O[d4][r], hh, ll);
                    ctx_hi[idx] = hh;
                    ctx_lo[idx] = ll;
                }
            }
        }
    }
}

// ---------------------------------------------------------------------------
extern "C" void kernel_launch(void* const* d_in, const int* in_sizes, int n_in,
                              void* d_out, int out_size, void* d_ws, size_t ws_size,
                              hipStream_t stream) {
    (void)in_sizes; (void)n_in; (void)out_size; (void)ws_size;

    const float* x      = (const float*)d_in[0];   // [64,256,2048]
    const float* w_qkv  = (const float*)d_in[1];   // [6144,2048]
    const float* w_proj = (const float*)d_in[2];   // [2048,2048]
    const float* b_proj = (const float*)d_in[3];   // [2048]
    float*       out    = (float*)d_out;           // [64,256,2048]

    // Workspace layout (all ushort planes; hi/lo pair costs same as one fp32)
    const size_t QKV_E   = (size_t)MTOT * QKV_N;   // 100663296 (q + kT + vT)
    const size_t CTX_E   = (size_t)MTOT * DIMC;    //  33554432
    const size_t WQKV_E  = (size_t)QKV_N * DIMC;   //  12582912
    const size_t WPROJ_E = (size_t)DIMC * DIMC;    //   4194304

    unsigned short* w       = (unsigned short*)d_ws;
    unsigned short* qkv_hi  = w;                       w += QKV_E;
    unsigned short* qkv_lo  = w;                       w += QKV_E;
    unsigned short* ctx_hi  = w;                       w += CTX_E;
    unsigned short* ctx_lo  = w;                       w += CTX_E;
    unsigned short* wqkv_hi = w;                       w += WQKV_E;
    unsigned short* wqkv_lo = w;                       w += WQKV_E;
    unsigned short* wprj_hi = w;                       w += WPROJ_E;
    unsigned short* wprj_lo = w;                       w += WPROJ_E;

    // 1) Split weights into bf16 hi/lo planes (w_qkv planes fit in 192MB L2).
    split_f32_kernel<<<dim3((unsigned)(WQKV_E / 256)), dim3(256), 0, stream>>>(
        w_qkv, wqkv_hi, wqkv_lo, (int)WQKV_E);
    split_f32_kernel<<<dim3((unsigned)(WPROJ_E / 256)), dim3(256), 0, stream>>>(
        w_proj, wprj_hi, wprj_lo, (int)WPROJ_E);

    // 2) qkv = x @ w_qkv^T -> q row-major + kT/vT transposed planes
    gemm_bf16x3_kernel<true, 0>
        <<<dim3(QKV_N / 128, MTOT / 128), dim3(256), 0, stream>>>(
            x, nullptr, nullptr, wqkv_hi, wqkv_lo, nullptr,
            nullptr, qkv_hi, qkv_lo, MTOT, QKV_N, DIMC);

    // 3) attention: (batch*head) x 2 j-halves, 4 waves each
    attn_kernel<<<dim3(BATCH * NHEAD * 2), dim3(128), 0, stream>>>(
        qkv_hi, qkv_lo, ctx_hi, ctx_lo);

    // 4) out = ctx @ w_proj^T + b_proj  (fp32)  M=16384 N=2048 K=2048
    gemm_bf16x3_kernel<false, 1>
        <<<dim3(DIMC / 128, MTOT / 128), dim3(256), 0, stream>>>(
            nullptr, ctx_hi, ctx_lo, wprj_hi, wprj_lo, b_proj,
            out, nullptr, nullptr, MTOT, DIMC, DIMC);
}
